// Spatial_MSI_64836826300480
// MI455X (gfx1250) — matrix operand order repaired and checked by reading, not compiled, not run on hardware
//
#include <hip/hip_runtime.h>
#include <hip/hip_bf16.h>
#include <type_traits>

// ---------------------------------------------------------------------------
// Spatial-MSI GAT autoencoder for MI455X (gfx1250, wave32, WMMA)
// Heavy GEMMs run on v_wmma_f32_16x16x32_f16 (f16 in, f32 accumulate).
// f32 -> f16 conversion fused into GEMM tile staging (single HBM read).
// f16-sourced A tiles (decoder GEMM) are staged with
// GLOBAL_LOAD_ASYNC_TO_LDS_B128 + s_wait_asynccnt (CDNA5 async path).
// ---------------------------------------------------------------------------

typedef _Float16 half8 __attribute__((ext_vector_type(8)));
typedef _Float16 v16h  __attribute__((ext_vector_type(16)));
typedef float    v8f   __attribute__((ext_vector_type(8)));

#define N_NODES 50000
#define N_EDGES 400000
#define IN_DIM  3000
#define HID     512
#define OUT_DIM 30
#define SLOPE   0.2f

// ======================= WMMA tiled GEMM ===================================
// C[M,Nn] = A[M,K] @ B[K,Nn] (+ bias[Nn]). Block tile 128x128, BK=32,
// 256 threads = 8 waves arranged 4(M) x 2(N); each wave owns 32x64
// (2x4 WMMA 16x16 tiles, f32 accumulators). Full K-tiles take an
// unguarded vector path; the single partial K-tile runs once after.
template <typename TA, typename TB>
__global__ __launch_bounds__(256) void gemm_wmma(
    const TA* __restrict__ A, int lda,
    const TB* __restrict__ B, int ldb,
    float* __restrict__ C, int ldc,
    const float* __restrict__ bias,
    int M, int Nn, int K)
{
    constexpr int BM = 128, BN = 128, BK = 32;
    constexpr int LDA_S = 40;   // LDS row stride (halves), 16B aligned
    constexpr int LDB_S = 40;   // LDS col stride for transposed B
    constexpr bool ASYNC_A = std::is_same<TA, _Float16>::value;

    __shared__ __align__(16) _Float16 sA[BM * LDA_S];   // 10 KB
    __shared__ __align__(16) _Float16 sBt[BN * LDB_S];  // 10 KB

    const int tid  = threadIdx.x;
    const int wave = tid >> 5, lane = tid & 31;
    const int wm   = wave >> 1, wn = wave & 1;       // 4x2 wave grid
    const int half = lane >> 4, l15 = lane & 15;
    const int row0 = blockIdx.y * BM, col0 = blockIdx.x * BN;

    v8f acc[2][4];
    const v8f vzero = {0.f, 0.f, 0.f, 0.f, 0.f, 0.f, 0.f, 0.f};
#pragma unroll
    for (int i = 0; i < 2; ++i)
#pragma unroll
        for (int j = 0; j < 4; ++j) acc[i][j] = vzero;

    const int ar   = tid >> 1;        // A row 0..127
    const int akc  = (tid & 1) * 16;  // A k-chunk 0 / 16
    const int grow = row0 + ar;
    const bool rowOK = grow < M;

    const int bkk = tid >> 4;         // B k-row 0..15 (pass adds +16)
    const int bcc = (tid & 15) * 8;   // B col chunk
    const int gcB = col0 + bcc;
    const bool colOK = gcB + 8 <= Nn; // Nn % 8 == 0 in all uses

    half8 hz;
#pragma unroll
    for (int i = 0; i < 8; ++i) hz[i] = (_Float16)0.f;

    // ---- staging: full (unguarded-in-K) K-tile ----
    auto stageA_full = [&](int k0) {
        const int kb = k0 + akc;
        if (rowOK) {
            if constexpr (ASYNC_A) {
                // pure byte copy: async DMA straight into LDS (ASYNCcnt)
                const TA* src = A + (size_t)grow * lda + kb;
                unsigned l0 = (unsigned)(uintptr_t)(&sA[ar * LDA_S + akc]);
                asm volatile(
                    "global_load_async_to_lds_b128 %0, %2, off\n\t"
                    "global_load_async_to_lds_b128 %1, %3, off"
                    :: "v"(l0), "v"(l0 + 16u), "v"(src), "v"(src + 8)
                    : "memory");
            } else {
                const float4* s4 =
                    reinterpret_cast<const float4*>(A + (size_t)grow * lda + kb);
                half8 h0, h1;
#pragma unroll
                for (int i = 0; i < 2; ++i) {
                    float4 v = s4[i];
                    h0[4*i+0] = (_Float16)v.x; h0[4*i+1] = (_Float16)v.y;
                    h0[4*i+2] = (_Float16)v.z; h0[4*i+3] = (_Float16)v.w;
                }
#pragma unroll
                for (int i = 0; i < 2; ++i) {
                    float4 v = s4[2 + i];
                    h1[4*i+0] = (_Float16)v.x; h1[4*i+1] = (_Float16)v.y;
                    h1[4*i+2] = (_Float16)v.z; h1[4*i+3] = (_Float16)v.w;
                }
                *reinterpret_cast<half8*>(&sA[ar * LDA_S + akc])     = h0;
                *reinterpret_cast<half8*>(&sA[ar * LDA_S + akc + 8]) = h1;
            }
        } else {
            *reinterpret_cast<half8*>(&sA[ar * LDA_S + akc])     = hz;
            *reinterpret_cast<half8*>(&sA[ar * LDA_S + akc + 8]) = hz;
        }
    };

    auto stageB_full = [&](int k0) {
#pragma unroll
        for (int p = 0; p < 2; ++p) {
            const int kk = bkk + p * 16;
            const int gk = k0 + kk;
            _Float16 tb[8];
            if (colOK) {
                const TB* src = B + (size_t)gk * ldb + gcB;
                if constexpr (std::is_same<TB, float>::value) {
                    const float4* s4 = reinterpret_cast<const float4*>(src);
                    float4 v0 = s4[0], v1 = s4[1];
                    tb[0]=(_Float16)v0.x; tb[1]=(_Float16)v0.y;
                    tb[2]=(_Float16)v0.z; tb[3]=(_Float16)v0.w;
                    tb[4]=(_Float16)v1.x; tb[5]=(_Float16)v1.y;
                    tb[6]=(_Float16)v1.z; tb[7]=(_Float16)v1.w;
                } else {
                    const half8* s8 = reinterpret_cast<const half8*>(src);
                    half8 v = s8[0];
#pragma unroll
                    for (int i = 0; i < 8; ++i) tb[i] = v[i];
                }
            } else {
#pragma unroll
                for (int i = 0; i < 8; ++i) tb[i] = (_Float16)0.f;
            }
#pragma unroll
            for (int i = 0; i < 8; ++i) sBt[(bcc + i) * LDB_S + kk] = tb[i];
        }
    };

    // ---- staging: partial (guarded) final K-tile ----
    auto stageA_tail = [&](int k0) {
        const int kb = k0 + akc;
        _Float16 tmp[16];
#pragma unroll
        for (int i = 0; i < 16; ++i) {
            int kk = kb + i;
            float v = (rowOK && kk < K) ? (float)A[(size_t)grow * lda + kk] : 0.f;
            tmp[i] = (_Float16)v;
        }
        half8 h0, h1;
#pragma unroll
        for (int i = 0; i < 8; ++i) { h0[i] = tmp[i]; h1[i] = tmp[8 + i]; }
        *reinterpret_cast<half8*>(&sA[ar * LDA_S + akc])     = h0;
        *reinterpret_cast<half8*>(&sA[ar * LDA_S + akc + 8]) = h1;
    };

    auto stageB_tail = [&](int k0) {
#pragma unroll
        for (int p = 0; p < 2; ++p) {
            const int kk = bkk + p * 16;
            const int gk = k0 + kk;
#pragma unroll
            for (int i = 0; i < 8; ++i) {
                float v = (gk < K && gcB + i < Nn)
                              ? (float)B[(size_t)gk * ldb + gcB + i] : 0.f;
                sBt[(bcc + i) * LDB_S + kk] = (_Float16)v;
            }
        }
    };

    // ---- fragments + 8 WMMAs (ISA 7.12.2 layouts) ----
    auto compute = [&]() {
        v16h aF[2];
#pragma unroll
        for (int rt = 0; rt < 2; ++rt) {
            const int r = wm * 32 + rt * 16 + l15;
            half8 lo = *reinterpret_cast<const half8*>(&sA[r * LDA_S + half * 8]);
            half8 hi = *reinterpret_cast<const half8*>(&sA[r * LDA_S + 16 + half * 8]);
            aF[rt] = __builtin_shufflevector(lo, hi,
                         0,1,2,3,4,5,6,7,8,9,10,11,12,13,14,15);
        }
#pragma unroll
        for (int ct = 0; ct < 4; ++ct) {
            const int c = wn * 64 + ct * 16 + l15;
            half8 lo = *reinterpret_cast<const half8*>(&sBt[c * LDB_S + half * 8]);
            half8 hi = *reinterpret_cast<const half8*>(&sBt[c * LDB_S + 16 + half * 8]);
            v16h bF = __builtin_shufflevector(lo, hi,
                          0,1,2,3,4,5,6,7,8,9,10,11,12,13,14,15);
#pragma unroll
            for (int rt = 0; rt < 2; ++rt)
                acc[rt][ct] = __builtin_amdgcn_wmma_f32_16x16x32_f16(
                    false, aF[rt], false, bF, (short)0, acc[rt][ct], false, false);
        }
    };

    const int kFull = K / BK;

    for (int t = 0; t < kFull; ++t) {
        const int k0 = t * BK;
        stageA_full(k0);
        stageB_full(k0);
        if constexpr (ASYNC_A)
            asm volatile("s_wait_asynccnt 0x0" ::: "memory");
        __syncthreads();

        if (t + 1 < kFull) {  // speculative prefetch of next tiles
            if (rowOK)
                __builtin_prefetch(A + (size_t)grow * lda + k0 + BK + akc, 0, 0);
            if (colOK)
                __builtin_prefetch(B + (size_t)(k0 + BK + bkk) * ldb + gcB, 0, 0);
        }

        compute();
        __syncthreads();
    }

    if (kFull * BK < K) {
        stageA_tail(kFull * BK);
        stageB_tail(kFull * BK);
        __syncthreads();
        compute();
        __syncthreads();
    }

    // ---- store: D layout VGPR r -> M = r + 8*half, N = lane&15 ----
#pragma unroll
    for (int rt = 0; rt < 2; ++rt)
#pragma unroll
        for (int ct = 0; ct < 4; ++ct) {
            const int gcol = col0 + wn * 64 + ct * 16 + l15;
            if (gcol >= Nn) continue;
            const float b = bias ? bias[gcol] : 0.f;
#pragma unroll
            for (int r = 0; r < 8; ++r) {
                const int grW = row0 + wm * 32 + rt * 16 + half * 8 + r;
                if (grW < M) C[(size_t)grW * ldc + gcol] = acc[rt][ct][r] + b;
            }
        }
}

// ======================= small kernels =====================================

__global__ void fill_f32(float* __restrict__ p, float v, long long n) {
    long long i = (long long)blockIdx.x * blockDim.x + threadIdx.x;
    if (i < n) p[i] = v;
}

// s_node = z @ a_src ; d_node = z @ a_dst  (wave per row)
__global__ __launch_bounds__(256) void rowdot2(
    const float* __restrict__ z, const float* __restrict__ a_src,
    const float* __restrict__ a_dst, float* __restrict__ s_node,
    float* __restrict__ d_node, int n)
{
    int row  = blockIdx.x * 8 + (threadIdx.x >> 5);
    int lane = threadIdx.x & 31;
    if (row >= n) return;
    float ps = 0.f, pd = 0.f;
    const float* zr = z + (size_t)row * HID;
    for (int k = lane; k < HID; k += 32) {
        float v = zr[k];
        ps += v * a_src[k];
        pd += v * a_dst[k];
    }
#pragma unroll
    for (int o = 16; o > 0; o >>= 1) {
        ps += __shfl_down(ps, o, 32);
        pd += __shfl_down(pd, o, 32);
    }
    if (lane == 0) { s_node[row] = ps; d_node[row] = pd; }
}

__device__ inline float lrelu(float a) { return a > 0.f ? a : SLOPE * a; }

__device__ inline void atomicMaxF(float* addr, float val) {
    // IEEE monotone-bit trick; mixed-sign safe with -inf init
    if (val >= 0.f) atomicMax((int*)addr, __float_as_int(val));
    else            atomicMin((unsigned int*)addr, __float_as_uint(val));
}

__global__ void edge_max(const int* __restrict__ src, const int* __restrict__ dst,
                         const float* __restrict__ sn, const float* __restrict__ dn,
                         float* __restrict__ m, int e)
{
    int i = blockIdx.x * blockDim.x + threadIdx.x;
    if (i >= e) return;
    int s = src[i], d = dst[i];
    atomicMaxF(&m[d], lrelu(sn[s] + dn[d]));
}

__global__ void edge_expsum(const int* __restrict__ src, const int* __restrict__ dst,
                            const float* __restrict__ sn, const float* __restrict__ dn,
                            const float* __restrict__ m, float* __restrict__ ssum, int e)
{
    int i = blockIdx.x * blockDim.x + threadIdx.x;
    if (i >= e) return;
    int s = src[i], d = dst[i];
    atomicAdd(&ssum[d], __expf(lrelu(sn[s] + dn[d]) - m[d]));
}

// agg[dst] += softmax_weight * z[src]   (one wave per edge, 512-wide)
__global__ __launch_bounds__(256) void edge_scatter(
    const int* __restrict__ src, const int* __restrict__ dst,
    const float* __restrict__ sn, const float* __restrict__ dn,
    const float* __restrict__ m, const float* __restrict__ ssum,
    const float* __restrict__ z, float* __restrict__ agg, int e)
{
    int i = blockIdx.x * 8 + (threadIdx.x >> 5);
    if (i >= e) return;
    int lane = threadIdx.x & 31;
    int s = src[i], d = dst[i];
    float w = __expf(lrelu(sn[s] + dn[d]) - m[d]) / ssum[d];
    const float* zr = z + (size_t)s * HID;
    float* ag = agg + (size_t)d * HID;
    for (int k = lane; k < HID; k += 32) atomicAdd(&ag[k], w * zr[k]);
}

// h2 = elu(agg) @ W2   (wave per row, ELU'd row staged in LDS)
__global__ __launch_bounds__(256) void h2_kernel(
    const float* __restrict__ agg, const float* __restrict__ W2,
    float* __restrict__ h2, int n)
{
    __shared__ float srow[8][HID];
    int wave = threadIdx.x >> 5, lane = threadIdx.x & 31;
    int row = blockIdx.x * 8 + wave;
    if (row < n) {
        const float* ar = agg + (size_t)row * HID;
        for (int k = lane; k < HID; k += 32) {
            float x = ar[k];
            srow[wave][k] = x > 0.f ? x : (__expf(x) - 1.f);
        }
    }
    __syncthreads();
    if (row >= n || lane >= OUT_DIM) return;
    float acc = 0.f;
    for (int k = 0; k < HID; ++k) acc += srow[wave][k] * W2[k * OUT_DIM + lane];
    h2[(size_t)row * OUT_DIM + lane] = acc;
}

// rec1h = f16( elu(h2 @ Wd1 + bd1) )   (block per row)
__global__ __launch_bounds__(512) void dec1_kernel(
    const float* __restrict__ h2, const float* __restrict__ Wd1,
    const float* __restrict__ bd1, _Float16* __restrict__ rec1h, int n)
{
    __shared__ float sh[OUT_DIM];
    int row = blockIdx.x;
    if (row >= n) return;
    int c = threadIdx.x;
    if (c < OUT_DIM) sh[c] = h2[(size_t)row * OUT_DIM + c];
    __syncthreads();
    float acc = bd1[c];
#pragma unroll
    for (int k = 0; k < OUT_DIM; ++k) acc += sh[k] * Wd1[k * HID + c];
    float e = acc > 0.f ? acc : (__expf(acc) - 1.f);
    rec1h[(size_t)row * HID + c] = (_Float16)e;
}

// CSL: sums[src] += h2[dst]; cnt[src] += 1   (wave per edge)
__global__ __launch_bounds__(256) void csl_edge(
    const int* __restrict__ src, const int* __restrict__ dst,
    const float* __restrict__ h2, float* __restrict__ sums,
    float* __restrict__ cnt, int e)
{
    int i = blockIdx.x * 8 + (threadIdx.x >> 5);
    if (i >= e) return;
    int lane = threadIdx.x & 31;
    int s = src[i], d = dst[i];
    if (lane < OUT_DIM) atomicAdd(&sums[(size_t)s * OUT_DIM + lane],
                                  h2[(size_t)d * OUT_DIM + lane]);
    if (lane == 0) atomicAdd(&cnt[s], 1.f);
}

__global__ void csl_div(float* __restrict__ sums, const float* __restrict__ cnt, int n)
{
    int i = blockIdx.x * blockDim.x + threadIdx.x;
    if (i >= n * OUT_DIM) return;
    sums[i] /= fmaxf(cnt[i / OUT_DIM], 1.f);
}

// ======================= launcher ==========================================

extern "C" void kernel_launch(void* const* d_in, const int* in_sizes, int n_in,
                              void* d_out, int out_size, void* d_ws, size_t ws_size,
                              hipStream_t stream)
{
    const float* features = (const float*)d_in[0];
    const int*   eidx     = (const int*)d_in[1];   // [2,E]: src = +0, dst = +E
    const int*   ecsl     = (const int*)d_in[2];
    const float* W1  = (const float*)d_in[3];
    const float* a1s = (const float*)d_in[4];
    const float* a1d = (const float*)d_in[5];
    const float* W2  = (const float*)d_in[6];
    const float* Wd1 = (const float*)d_in[7];
    const float* bd1 = (const float*)d_in[8];
    const float* Wd2 = (const float*)d_in[9];
    const float* bd2 = (const float*)d_in[10];

    float* out   = (float*)d_out;
    float* h2    = out;
    float* hpos  = out + (size_t)N_NODES * OUT_DIM;
    float* h2neg = out + 2 * (size_t)N_NODES * OUT_DIM;
    float* rec   = out + 3 * (size_t)N_NODES * OUT_DIM;

    char* ws = (char*)d_ws;
    float* z   = (float*)ws;            ws += (size_t)N_NODES * HID * sizeof(float);
    float* agg = (float*)ws;            ws += (size_t)N_NODES * HID * sizeof(float);
    _Float16* rec1h = (_Float16*)ws;    ws += (size_t)N_NODES * HID * sizeof(_Float16);
    float* s_node = (float*)ws;         ws += (size_t)N_NODES * sizeof(float);
    float* d_node = (float*)ws;         ws += (size_t)N_NODES * sizeof(float);
    float* m_buf  = (float*)ws;         ws += (size_t)N_NODES * sizeof(float);
    float* ssum   = (float*)ws;         ws += (size_t)N_NODES * sizeof(float);
    float* cnt    = (float*)ws;         ws += (size_t)N_NODES * sizeof(float);

    const int E = N_EDGES, N = N_NODES;
    const float ninf = -__builtin_inff();

    dim3 blk256(256), blk512(512);
    dim3 gE1((E + 255) / 256);       // thread-per-edge
    dim3 gE8((E + 7) / 8);           // wave-per-edge
    dim3 gN8((N + 7) / 8);           // wave-per-row
    dim3 gGemm1(HID / 128, (N + 127) / 128);            // (4, 391)
    dim3 gGemm3((IN_DIM + 127) / 128, (N + 127) / 128); // (24, 391)

    // 1) z = features @ W1  (fp32 sources -> f16 WMMA, f32 accumulate)
    gemm_wmma<float, float><<<gGemm1, blk256, 0, stream>>>(
        features, IN_DIM, W1, HID, z, HID, nullptr, N, HID, IN_DIM);

    // 2) per-node attention scalars (shared by pos + neg passes)
    rowdot2<<<gN8, blk256, 0, stream>>>(z, a1s, a1d, s_node, d_node, N);

    // 3) GAT aggregation, positive edges -> h2
    fill_f32<<<(N + 255) / 256, blk256, 0, stream>>>(m_buf, ninf, N);
    fill_f32<<<(N + 255) / 256, blk256, 0, stream>>>(ssum, 0.f, N);
    fill_f32<<<((long long)N * HID + 255) / 256, blk256, 0, stream>>>(agg, 0.f, (long long)N * HID);
    edge_max    <<<gE1, blk256, 0, stream>>>(eidx, eidx + E, s_node, d_node, m_buf, E);
    edge_expsum <<<gE1, blk256, 0, stream>>>(eidx, eidx + E, s_node, d_node, m_buf, ssum, E);
    edge_scatter<<<gE8, blk256, 0, stream>>>(eidx, eidx + E, s_node, d_node, m_buf, ssum, z, agg, E);
    h2_kernel   <<<gN8, blk256, 0, stream>>>(agg, W2, h2, N);

    // 4) GAT aggregation, negative (CSL) edges -> h2_neg (reuse agg/m/ssum)
    fill_f32<<<(N + 255) / 256, blk256, 0, stream>>>(m_buf, ninf, N);
    fill_f32<<<(N + 255) / 256, blk256, 0, stream>>>(ssum, 0.f, N);
    fill_f32<<<((long long)N * HID + 255) / 256, blk256, 0, stream>>>(agg, 0.f, (long long)N * HID);
    edge_max    <<<gE1, blk256, 0, stream>>>(ecsl, ecsl + E, s_node, d_node, m_buf, E);
    edge_expsum <<<gE1, blk256, 0, stream>>>(ecsl, ecsl + E, s_node, d_node, m_buf, ssum, E);
    edge_scatter<<<gE8, blk256, 0, stream>>>(ecsl, ecsl + E, s_node, d_node, m_buf, ssum, z, agg, E);
    h2_kernel   <<<gN8, blk256, 0, stream>>>(agg, W2, h2neg, N);

    // 5) decoder stage 1: rec1h = f16(elu(h2 @ Wd1 + bd1))
    dec1_kernel<<<N, blk512, 0, stream>>>(h2, Wd1, bd1, rec1h, N);

    // 6) decoder stage 2 (big GEMM, async-LDS A path): rec = rec1h @ Wd2 + bd2
    gemm_wmma<_Float16, float><<<gGemm3, blk256, 0, stream>>>(
        rec1h, HID, Wd2, IN_DIM, rec, IN_DIM, bd2, N, IN_DIM, HID);

    // 7) CSL scatter-mean of h2 over positive edges -> h_pos
    fill_f32<<<((long long)N * OUT_DIM + 255) / 256, blk256, 0, stream>>>(hpos, 0.f, (long long)N * OUT_DIM);
    fill_f32<<<(N + 255) / 256, blk256, 0, stream>>>(cnt, 0.f, N);
    csl_edge<<<gE8, blk256, 0, stream>>>(eidx, eidx + E, h2, hpos, cnt, E);
    csl_div <<<((N * OUT_DIM) + 255) / 256, blk256, 0, stream>>>(hpos, cnt, N);
}
